// fuse_block_CA_v2_75402445849292
// MI455X (gfx1250) — matrix operand order repaired and checked by reading, not compiled, not run on hardware
//
#include <hip/hip_runtime.h>

#define B_   8
#define CH_  512
#define HW_  6400
#define K_   1024
#define CR_  32

typedef _Float16 v16h __attribute__((ext_vector_type(16)));
typedef _Float16 v8h  __attribute__((ext_vector_type(8)));
typedef float    v8f  __attribute__((ext_vector_type(8)));

union V16 { v16h v; v8h h[2]; };

// ---------------- K0: w_fuse f32 -> f16 ----------------
__global__ void k_convert_w(const float* __restrict__ w, _Float16* __restrict__ w16) {
  int i = (blockIdx.x * 256 + threadIdx.x) * 4;           // 512 blocks -> 524288
  float4 f = *(const float4*)(w + i);
  w16[i + 0] = (_Float16)f.x;
  w16[i + 1] = (_Float16)f.y;
  w16[i + 2] = (_Float16)f.z;
  w16[i + 3] = (_Float16)f.w;
}

// ---------------- K1: per-(b,c) row/col means of rgb and t ----------------
__global__ void k_means(const float* __restrict__ rgb, const float* __restrict__ tt,
                        float* __restrict__ rgb_h, float* __restrict__ rgb_w,
                        float* __restrict__ t_h,   float* __restrict__ t_w) {
  __shared__ float sh[80];
  __shared__ float sw[80];
  int bid = blockIdx.x;                 // 8192 blocks: b * 1024 + cc
  int b  = bid >> 10;
  int cc = bid & 1023;
  const float* src = (cc < CH_)
      ? rgb + (size_t)(b * CH_ + cc) * HW_
      : tt  + (size_t)(b * CH_ + (cc - CH_)) * HW_;
  int tid = threadIdx.x;
  if (tid < 80) { sh[tid] = 0.f; sw[tid] = 0.f; }
  __syncthreads();
  for (int i = tid; i < HW_; i += 256) {
    float v = src[i];
    int h = i / 80;
    int w = i - h * 80;
    atomicAdd(&sh[h], v);
    atomicAdd(&sw[w], v);
  }
  __syncthreads();
  if (tid < 80) {
    float mh = sh[tid] * (1.f / 80.f);
    float mw = sw[tid] * (1.f / 80.f);
    size_t o = (size_t)(b * CH_ + (cc & (CH_ - 1))) * 80 + tid;
    if (cc < CH_) { rgb_h[o] = mh; rgb_w[o] = mw; }
    else          { t_h[o]   = mh; t_w[o]   = mw; }
  }
}

// ---------------- K2a: y = BN(ReLU applied after)(w_r1 @ cat(rgb_h,rgb_w)), z = ReLU(w_t1 @ cat(t_h,t_w)) ----
__global__ void k_small1(const float* __restrict__ rgb_h, const float* __restrict__ rgb_w,
                         const float* __restrict__ t_h,   const float* __restrict__ t_w,
                         const float* __restrict__ w_r1,  const float* __restrict__ w_t1,
                         const float* __restrict__ gma,   const float* __restrict__ bta,
                         const float* __restrict__ mu,    const float* __restrict__ var,
                         float* __restrict__ yws, float* __restrict__ zws) {
  int e = blockIdx.x * 256 + threadIdx.x;   // 320 blocks -> 81920 = 8*2*32*160
  int col = e % 160;
  int t1  = e / 160;
  int o   = t1 & 31;
  int t2  = t1 >> 5;
  int mat = t2 & 1;
  int b   = t2 >> 1;
  const float* wrow = (mat ? w_t1 : w_r1) + o * CH_;
  const float* colsrc;
  int x;
  if (col < 80) { colsrc = (mat ? t_h : rgb_h); x = col; }
  else          { colsrc = (mat ? t_w : rgb_w); x = col - 80; }
  colsrc += (size_t)b * CH_ * 80 + x;
  float acc = 0.f;
  #pragma unroll 8
  for (int c = 0; c < CH_; ++c) acc += wrow[c] * colsrc[(size_t)c * 80];
  if (mat == 0) {
    float rs = rsqrtf(var[o] + 1e-5f);
    acc = acc * (gma[o] * rs) + (bta[o] - gma[o] * mu[o] * rs);
  }
  acc = fmaxf(acc, 0.f);
  (mat ? zws : yws)[(b * CR_ + o) * 160 + col] = acc;
}

// ---------------- K2b: the four sigmoid gates ----------------
__global__ void k_small2(const float* __restrict__ yws, const float* __restrict__ zws,
                         const float* __restrict__ w_fh1, const float* __restrict__ w_fw1,
                         const float* __restrict__ w_fh2, const float* __restrict__ w_fw2,
                         float* __restrict__ s_h1, float* __restrict__ s_w1,
                         float* __restrict__ s_h2, float* __restrict__ s_w2) {
  int e = blockIdx.x * 256 + threadIdx.x;   // 5120 blocks -> 1310720 = 8*4*512*80
  int x  = e % 80;
  int t1 = e / 80;
  int o  = t1 & (CH_ - 1);
  int t2 = t1 >> 9;
  int arr = t2 & 3;
  int b   = t2 >> 2;
  const float* wr; const float* src; float* dst; int coff;
  switch (arr) {
    case 0:  wr = w_fh1 + o * CR_; src = yws; coff = x;      dst = s_h1; break;
    case 1:  wr = w_fw1 + o * CR_; src = yws; coff = 80 + x; dst = s_w1; break;
    case 2:  wr = w_fh2 + o * CR_; src = zws; coff = x;      dst = s_h2; break;
    default: wr = w_fw2 + o * CR_; src = zws; coff = 80 + x; dst = s_w2; break;
  }
  float acc = 0.f;
  #pragma unroll
  for (int j = 0; j < CR_; ++j) acc += wr[j] * src[(b * CR_ + j) * 160 + coff];
  dst[(size_t)(b * CH_ + o) * 80 + x] = 1.f / (1.f + expf(-acc));
}

// ---------------- K3: fused WMMA GEMM + rank-1 gating epilogue ----------------
// grid = (HW/64, CH/128, B), block = 256 (8 wave32)
#define LDA 40   // padded f16 row stride (80B) -> conflict-free b128 fragment loads
#define LDB 40
__global__ void __launch_bounds__(256)
k_gemm(const float* __restrict__ rgb, const float* __restrict__ tt,
       const _Float16* __restrict__ w16,
       const float* __restrict__ s_h1, const float* __restrict__ s_w1,
       const float* __restrict__ s_h2, const float* __restrict__ s_w2,
       float* __restrict__ out) {
  __shared__ _Float16 sA[128 * LDA];   // [m][k] weights tile, padded
  __shared__ _Float16 sB[64 * LDB];    // [n][k] activations tile (transposed), padded

  const int tid  = threadIdx.x;
  const int lane = tid & 31;
  const int wv   = tid >> 5;
  const int wv_m = wv & 3;        // 4 wave-rows  -> 32 M each
  const int wv_n = wv >> 2;       // 2 wave-cols  -> 32 N each
  const int lm = lane & 15;
  const int lh = lane >> 4;

  const int n_blk = blockIdx.x * 64;
  const int o_blk = blockIdx.y * 128;
  const int bz    = blockIdx.z;

  v8f acc[2][2] = {};

  // B staging roles: thread t -> (k = t>>3, 8 consecutive n at g = t&7)
  const int kb_row = tid >> 3;
  const int gcol   = tid & 7;

  for (int kk = 0; kk < K_ / 32; ++kk) {
    const int k0 = kk * 32;
    __syncthreads();

    // ---- stage A: 128 x 32 f16 from w16 (row-major) ----
    #pragma unroll
    for (int rep = 0; rep < 2; ++rep) {
      int idx = tid + rep * 256;            // 512 segments of 8 halves
      int r = idx >> 2, q = idx & 3;
      v8h seg = *(const v8h*)(w16 + (size_t)(o_blk + r) * K_ + k0 + q * 8);
      *(v8h*)(sA + r * LDA + q * 8) = seg;
    }

    // ---- stage B: 32 x 64 f32 from rgb/t, convert + transpose to [n][k] ----
    {
      int kg = k0 + kb_row;
      const float* row = (kg < CH_)
          ? rgb + (size_t)(bz * CH_ + kg) * HW_ + n_blk
          : tt  + (size_t)(bz * CH_ + (kg - CH_)) * HW_ + n_blk;
      const float4* p = (const float4*)(row + gcol * 8);
      float4 f0 = p[0], f1 = p[1];
      float vv[8] = {f0.x, f0.y, f0.z, f0.w, f1.x, f1.y, f1.z, f1.w};
      #pragma unroll
      for (int u = 0; u < 8; ++u)
        sB[(gcol * 8 + u) * LDB + kb_row] = (_Float16)vv[u];
    }
    __syncthreads();

    // ---- fragments + 4 WMMAs per wave ----
    V16 a0, a1, b0, b1;
    const int klo = lh * 8;                  // A: lanes0-15 K{0..7,16..23}, lanes16-31 K{8..15,24..31}
    const int r0  = wv_m * 32 + lm;
    a0.h[0] = *(const v8h*)(sA + r0 * LDA + klo);
    a0.h[1] = *(const v8h*)(sA + r0 * LDA + klo + 16);
    const int r1 = r0 + 16;
    a1.h[0] = *(const v8h*)(sA + r1 * LDA + klo);
    a1.h[1] = *(const v8h*)(sA + r1 * LDA + klo + 16);

    const int kbf = lh * 8;                  // B: lanes0-15 K{0..7,16..23}, lanes16-31 K{8..15,24..31}
    const int c0  = wv_n * 32 + lm;
    b0.h[0] = *(const v8h*)(sB + c0 * LDB + kbf);
    b0.h[1] = *(const v8h*)(sB + c0 * LDB + kbf + 16);
    const int c1 = c0 + 16;
    b1.h[0] = *(const v8h*)(sB + c1 * LDB + kbf);
    b1.h[1] = *(const v8h*)(sB + c1 * LDB + kbf + 16);

    acc[0][0] = __builtin_amdgcn_wmma_f32_16x16x32_f16(false, a0.v, false, b0.v, (short)0, acc[0][0], false, false);
    acc[0][1] = __builtin_amdgcn_wmma_f32_16x16x32_f16(false, a0.v, false, b1.v, (short)0, acc[0][1], false, false);
    acc[1][0] = __builtin_amdgcn_wmma_f32_16x16x32_f16(false, a1.v, false, b0.v, (short)0, acc[1][0], false, false);
    acc[1][1] = __builtin_amdgcn_wmma_f32_16x16x32_f16(false, a1.v, false, b1.v, (short)0, acc[1][1], false, false);
  }

  // ---- epilogue: out = c * (s_h1*s_w1 + s_h2*s_w2) ----
  #pragma unroll
  for (int i = 0; i < 2; ++i) {
    #pragma unroll
    for (int j = 0; j < 2; ++j) {
      int obase = o_blk + wv_m * 32 + i * 16 + lh * 8;   // C layout: lane>=16 -> M+8
      int n     = n_blk + wv_n * 32 + j * 16 + lm;       // N = lane&15
      int h = n / 80;
      int w = n - h * 80;
      #pragma unroll
      for (int v = 0; v < 8; ++v) {
        int o = obase + v;
        size_t so = (size_t)(bz * CH_ + o) * 80;
        float sc = s_h1[so + h] * s_w1[so + w] + s_h2[so + h] * s_w2[so + w];
        out[(size_t)(bz * CH_ + o) * HW_ + n] = acc[i][j][v] * sc;
      }
    }
  }
}

// ---------------- launch ----------------
extern "C" void kernel_launch(void* const* d_in, const int* in_sizes, int n_in,
                              void* d_out, int out_size, void* d_ws, size_t ws_size,
                              hipStream_t stream) {
  const float* rgb    = (const float*)d_in[0];
  const float* tt     = (const float*)d_in[1];
  const float* w_fuse = (const float*)d_in[2];
  const float* w_r1   = (const float*)d_in[3];
  const float* w_t1   = (const float*)d_in[4];
  const float* w_fh1  = (const float*)d_in[5];
  const float* w_fw1  = (const float*)d_in[6];
  const float* w_fh2  = (const float*)d_in[7];
  const float* w_fw2  = (const float*)d_in[8];
  const float* gma    = (const float*)d_in[9];
  const float* bta    = (const float*)d_in[10];
  const float* mu     = (const float*)d_in[11];
  const float* var    = (const float*)d_in[12];
  float* out = (float*)d_out;

  char* ws = (char*)d_ws;
  size_t off = 0;
  _Float16* w16 = (_Float16*)(ws + off); off += (size_t)CH_ * K_ * 2;          // 1 MB
  float* rgb_h  = (float*)(ws + off);    off += (size_t)B_ * CH_ * 80 * 4;
  float* rgb_w  = (float*)(ws + off);    off += (size_t)B_ * CH_ * 80 * 4;
  float* t_h    = (float*)(ws + off);    off += (size_t)B_ * CH_ * 80 * 4;
  float* t_w    = (float*)(ws + off);    off += (size_t)B_ * CH_ * 80 * 4;
  float* s_h1   = (float*)(ws + off);    off += (size_t)B_ * CH_ * 80 * 4;
  float* s_w1   = (float*)(ws + off);    off += (size_t)B_ * CH_ * 80 * 4;
  float* s_h2   = (float*)(ws + off);    off += (size_t)B_ * CH_ * 80 * 4;
  float* s_w2   = (float*)(ws + off);    off += (size_t)B_ * CH_ * 80 * 4;
  float* yws    = (float*)(ws + off);    off += (size_t)B_ * CR_ * 160 * 4;
  float* zws    = (float*)(ws + off);    off += (size_t)B_ * CR_ * 160 * 4;   // ~11.9 MB total

  k_convert_w<<<512, 256, 0, stream>>>(w_fuse, w16);
  k_means<<<B_ * 2 * CH_ / 1 / 1, 256, 0, stream>>>(rgb, tt, rgb_h, rgb_w, t_h, t_w); // 8192 blocks
  k_small1<<<320, 256, 0, stream>>>(rgb_h, rgb_w, t_h, t_w, w_r1, w_t1,
                                    gma, bta, mu, var, yws, zws);
  k_small2<<<5120, 256, 0, stream>>>(yws, zws, w_fh1, w_fw1, w_fh2, w_fw2,
                                     s_h1, s_w1, s_h2, s_w2);
  dim3 g3(HW_ / 64, CH_ / 128, B_);
  k_gemm<<<g3, 256, 0, stream>>>(rgb, tt, w16, s_h1, s_w1, s_h2, s_w2, out);
}
